// Encoder_MPNN_84731114815923
// MI455X (gfx1250) — matrix operand order repaired and checked by reading, not compiled, not run on hardware
//
#include <hip/hip_runtime.h>
#include <hip/hip_bf16.h>
#include <math.h>

// ---------------------------------------------------------------------------
// MI455X (gfx1250) implementation of the MPNN encoder reference.
//
// All heavy GEMMs (edge MLPs, edge embedding, node FFNs) run on
// v_wmma_f32_16x16x32_bf16 (wave32 WMMA, f32 accumulate). Operands are staged
// in LDS as bf16 with the exact CDNA5 WMMA VGPR striping; weights are
// pre-converted once per launch to bf16 row-major (K x N), which is exactly
// the B-operand layout.
//
// Assumption (documented): the harness flattens the `params` pytree into
// separate d_in leaves in JAX pytree order (dict keys sorted lexicographically,
// lists in order, lin = {b,w}, ln = {b,g}), inserted at the position of
// 'params' in setup_inputs() dict order. That yields:
//   d_in[0]=trans_1, d_in[1]=rotmats_1, d_in[2..123]=122 param leaves,
//   d_in[124]=motif_mask, [125]=residue_mask, [126]=aatype,
//   [127]=residue_index, [128]=chain_index.
//
// jax.random.normal(key 42) (threefry) is approximated with a counter-hash
// Box-Muller (exact PRNG stream reproduction is out of scope).
// ---------------------------------------------------------------------------

#define BATCH   2
#define NRES    2048
#define NNODE   (BATCH * NRES)   // 4096
#define KNB     48
#define HIDDEN  128
#define LATENT  32

typedef __bf16 bf16;
typedef __attribute__((ext_vector_type(16))) __bf16 v16bf;
typedef __attribute__((ext_vector_type(8)))  float  v8f;

// -------------------------- WMMA fragment helpers --------------------------
// A-matrix 16x32 bf16, row-major source with leading dim lda (elements).
// Layout (ISA 7.12.2): lanes 0-15 = rows, elems 0..7 -> K = kh..kh+7,
// elems 8..15 -> K = 16+kh..16+kh+7 with kh = (lane>=16 ? 8 : 0).
__device__ inline v16bf wmma_load_a(const bf16* A, int lda) {
  const int lane = threadIdx.x & 31;
  const int m  = lane & 15;
  const int kh = (lane >> 4) * 8;
  const bf16* r = A + m * lda;
  v16bf a;
#pragma unroll
  for (int e = 0; e < 8; ++e) {
    a[e]     = r[kh + e];
    a[8 + e] = r[16 + kh + e];
  }
  return a;
}

// B-matrix 32x16 bf16, row-major K x N source with leading dim ldb.
// elems 0..7 -> K = kb..kb+7, elems 8..15 -> K = 16+kb..16+kb+7,
// kb = (lane>=16 ? 8 : 0). Column n = lane & 15.
__device__ inline v16bf wmma_load_b(const bf16* B, int ldb) {
  const int lane = threadIdx.x & 31;
  const int n  = lane & 15;
  const int kb = (lane >> 4) * 8;
  v16bf b;
#pragma unroll
  for (int e = 0; e < 8; ++e) {
    b[e]     = B[(size_t)(kb + e) * ldb + n];
    b[8 + e] = B[(size_t)(16 + kb + e) * ldb + n];
  }
  return b;
}

// C/D 16x16 f32: elem r -> row M = (lane>=16 ? 8 : 0) + r, col N = lane & 15.
__device__ inline float wmma_row(int r)  { return 0.f; }  // (placeholder, unused)

__device__ inline float fgelu(float x) {
  return 0.5f * x * (1.0f + erff(x * 0.70710678118654752f));
}

// Accumulate C += A(16xKT*32) * W(K x ldw) slice, per-wave col0 base.
template <int KT, int CT>
__device__ inline void gemm_tile(const bf16* A, int lda, const bf16* W, int ldw,
                                 int col0, v8f* acc) {
#pragma unroll
  for (int kt = 0; kt < KT; ++kt) {
    v16bf a = wmma_load_a(A + kt * 32, lda);
#pragma unroll
    for (int ct = 0; ct < CT; ++ct) {
      v16bf b = wmma_load_b(W + (size_t)(kt * 32) * ldw + col0 + ct * 16, ldw);
      acc[ct] = __builtin_amdgcn_wmma_f32_16x16x32_bf16(
          false, a, false, b, (short)0, acc[ct], false, false);
    }
  }
}

// Store one 16x16 accumulator into a bf16 row-major buffer with bias + gelu.
__device__ inline void epi_bf16_gelu(bf16* H, int ldh, int col0,
                                     const float* bias, v8f c) {
  const int lane = threadIdx.x & 31;
  const int n = lane & 15, mh = (lane >> 4) * 8;
#pragma unroll
  for (int r = 0; r < 8; ++r) {
    float x = c[r] + bias[col0 + n];
    H[(size_t)(mh + r) * ldh + col0 + n] = (bf16)fgelu(x);
  }
}

// Store one 16x16 accumulator into f32 row-major buffer with bias (no act).
__device__ inline void epi_f32(float* O, int ldh, int col0,
                               const float* bias, v8f c) {
  const int lane = threadIdx.x & 31;
  const int n = lane & 15, mh = (lane >> 4) * 8;
#pragma unroll
  for (int r = 0; r < 8; ++r)
    O[(size_t)(mh + r) * ldh + col0 + n] = c[r] + bias[col0 + n];
}

// ------------------------------- small utils -------------------------------
__global__ void cvt_f32_bf16_kernel(const float* __restrict__ s,
                                    bf16* __restrict__ d, int n) {
  int i = blockIdx.x * blockDim.x + threadIdx.x;
  if (i < n) d[i] = (bf16)s[i];
}

__global__ void zero_f32_kernel(float* __restrict__ p, int n) {
  int i = blockIdx.x * blockDim.x + threadIdx.x;
  if (i < n) p[i] = 0.f;
}

__device__ inline unsigned pcg_hash(unsigned x) {
  x = x * 747796405u + 2891336453u;
  x = ((x >> ((x >> 28u) + 4u)) ^ x) * 277803737u;
  return (x >> 22u) ^ x;
}

// ------------------------------ geometry stage -----------------------------
// atoms layout: atoms[node*15 + a*3 + xyz], a: 0=N 1=Ca 2=C 3=O 4=Cb
__global__ void atoms_kernel(const float* __restrict__ trans,
                             const float* __restrict__ rot,
                             const float* __restrict__ mm,
                             const float* __restrict__ rmask,
                             const int* __restrict__ resi,
                             const int* __restrict__ chain,
                             float* __restrict__ atoms,
                             int* __restrict__ ridx, int* __restrict__ cidx) {
  int n = blockIdx.x * blockDim.x + threadIdx.x;
  if (n >= NNODE) return;
  const float BB[4][3] = {{-0.5272f, 1.3593f, 0.f},
                          {0.f, 0.f, 0.f},
                          {1.5233f, 0.f, 0.f},
                          {2.1421f, -1.062f, 0.f}};
  float R[9], t[3];
#pragma unroll
  for (int i = 0; i < 9; ++i) R[i] = rot[(size_t)n * 9 + i];
#pragma unroll
  for (int i = 0; i < 3; ++i) t[i] = trans[(size_t)n * 3 + i];
  float msk = rmask[n] * mm[n];
  float X[4][3];
#pragma unroll
  for (int a = 0; a < 4; ++a)
#pragma unroll
    for (int i = 0; i < 3; ++i)
      X[a][i] = (t[i] + R[i * 3 + 0] * BB[a][0] + R[i * 3 + 1] * BB[a][1] +
                 R[i * 3 + 2] * BB[a][2]) * msk;
  float b[3], c[3], a3[3], Cb[3];
#pragma unroll
  for (int i = 0; i < 3; ++i) { b[i] = X[1][i] - X[0][i]; c[i] = X[2][i] - X[1][i]; }
  a3[0] = b[1] * c[2] - b[2] * c[1];
  a3[1] = b[2] * c[0] - b[0] * c[2];
  a3[2] = b[0] * c[1] - b[1] * c[0];
#pragma unroll
  for (int i = 0; i < 3; ++i)
    Cb[i] = -0.58273431f * a3[i] + 0.56802827f * b[i] - 0.54067466f * c[i] + X[1][i];
  float* o = atoms + (size_t)n * 15;
#pragma unroll
  for (int i = 0; i < 3; ++i) {
    o[0 + i] = X[0][i]; o[3 + i] = X[1][i]; o[6 + i] = X[2][i];
    o[9 + i] = X[3][i]; o[12 + i] = Cb[i];
  }
  ridx[n] = (int)((float)resi[n] * mm[n]);
  cidx[n] = (int)((float)chain[n] * mm[n]);
}

// ------------------------------- top-K (K=48) ------------------------------
__global__ __launch_bounds__(256) void topk_kernel(
    const float* __restrict__ atoms, const float* __restrict__ mm,
    int* __restrict__ Eidx) {
  const int node = blockIdx.x;
  const int base = (node >> 11) << 11;  // batch row start
  const int tid = threadIdx.x;
  __shared__ float dist[NRES];
  __shared__ float rv[256];
  __shared__ int ri[256];
  const float cax = atoms[(size_t)node * 15 + 3];
  const float cay = atoms[(size_t)node * 15 + 4];
  const float caz = atoms[(size_t)node * 15 + 5];
  const float mi = mm[node];
  float lmax = -1e30f;
  for (int j = tid; j < NRES; j += 256) {
    const float* g = atoms + (size_t)(base + j) * 15;
    float dx = cax - g[3], dy = cay - g[4], dz = caz - g[5];
    float d = sqrtf(dx * dx + dy * dy + dz * dz + 1e-6f);
    dist[j] = d;
    lmax = fmaxf(lmax, d);
  }
  rv[tid] = lmax;
  __syncthreads();
  for (int s = 128; s > 0; s >>= 1) {
    if (tid < s) rv[tid] = fmaxf(rv[tid], rv[tid + s]);
    __syncthreads();
  }
  float rowmax = rv[0];
  __syncthreads();
  for (int j = tid; j < NRES; j += 256) {
    float m2 = mi * mm[base + j];
    dist[j] += (1.f - m2) * rowmax;
  }
  __syncthreads();
  for (int k = 0; k < KNB; ++k) {
    float bv = 1e30f; int bi = 0x7fffffff;
    for (int j = tid; j < NRES; j += 256) {
      float v = dist[j];
      if (v < bv || (v == bv && j < bi)) { bv = v; bi = j; }
    }
    rv[tid] = bv; ri[tid] = bi;
    __syncthreads();
    for (int s = 128; s > 0; s >>= 1) {
      if (tid < s) {
        if (rv[tid + s] < rv[tid] ||
            (rv[tid + s] == rv[tid] && ri[tid + s] < ri[tid])) {
          rv[tid] = rv[tid + s]; ri[tid] = ri[tid + s];
        }
      }
      __syncthreads();
    }
    if (tid == 0) { Eidx[(size_t)node * KNB + k] = ri[0]; dist[ri[0]] = 1e30f; }
    __syncthreads();
  }
}

// -------------------- edge init: features -> edge_emb -> LN -> W_e ---------
__device__ __constant__ int PAIR_P[25] = {1,0,2,3,4,1,1,1,1,0,0,0,4,4,3,0,2,3,4,2,3,4,2,3,2};
__device__ __constant__ int PAIR_Q[25] = {1,0,2,3,4,0,2,3,4,2,3,4,2,3,2,1,1,1,1,0,0,0,4,4,3};

__global__ __launch_bounds__(128) void edge_init_kernel(
    const float* __restrict__ atoms, const int* __restrict__ ridx,
    const int* __restrict__ cidx, const int* __restrict__ Eidx,
    const float* __restrict__ posW, const float* __restrict__ posB,
    const bf16* __restrict__ Wemb, const float* __restrict__ bemb,
    const float* __restrict__ lng, const float* __restrict__ lnb,
    const bf16* __restrict__ We, const float* __restrict__ bWe,
    float* __restrict__ hE) {
  const int node = blockIdx.x;
  const int jb = (node >> 11) << 11;
  const int tid = threadIdx.x, wave = tid >> 5;
  __shared__ bf16 Fbuf[16 * 416];
  __shared__ float dbuf[16 * 25];
  __shared__ float ai[15];
  __shared__ bf16 Hbuf[16 * HIDDEN];
  __shared__ float Obuf[16 * HIDDEN];
  if (tid < 15) ai[tid] = atoms[(size_t)node * 15 + tid];
  const int ri = ridx[node], ci = cidx[node];
  __syncthreads();
  for (int mt = 0; mt < 3; ++mt) {
    const int e0 = node * KNB + mt * 16;
    for (int idx = tid; idx < 16 * 25; idx += 128) {
      int r = idx / 25, p = idx % 25;
      int jg = jb + Eidx[e0 + r];
      const float* aj = atoms + (size_t)jg * 15 + PAIR_Q[p] * 3;
      const float* ap = ai + PAIR_P[p] * 3;
      float dx = ap[0] - aj[0], dy = ap[1] - aj[1], dz = ap[2] - aj[2];
      dbuf[idx] = sqrtf(dx * dx + dy * dy + dz * dz + 1e-6f);
    }
    __syncthreads();
    for (int idx = tid; idx < 16 * 416; idx += 128) {
      int r = idx / 416, cc = idx % 416;
      float v;
      if (cc < 16) {
        int jg = jb + Eidx[e0 + r];
        int off = ri - ridx[jg];
        int dch = (ci == cidx[jg]) ? 1 : 0;
        int dcode = dch ? min(max(off + 32, 0), 64) : 65;
        v = posW[dcode * 16 + cc] + posB[cc];
      } else {
        int p = (cc - 16) >> 4, k = (cc - 16) & 15;
        float mu = 2.0f + 20.0f * (float)k / 15.0f;
        float t = (dbuf[r * 25 + p] - mu) * 0.8f;  // sigma = 1.25
        v = __expf(-t * t);
      }
      Fbuf[r * 416 + cc] = (bf16)v;
    }
    __syncthreads();
    {
      v8f acc[2] = {};
      gemm_tile<13, 2>(Fbuf, 416, Wemb, HIDDEN, wave * 32, acc);
      epi_f32(Obuf, HIDDEN, wave * 32 + 0, bemb, acc[0]);
      epi_f32(Obuf, HIDDEN, wave * 32 + 16, bemb, acc[1]);
    }
    __syncthreads();
    if (tid < 16) {
      const float* row = Obuf + tid * HIDDEN;
      float mean = 0.f;
      for (int c = 0; c < HIDDEN; ++c) mean += row[c];
      mean *= (1.f / HIDDEN);
      float var = 0.f;
      for (int c = 0; c < HIDDEN; ++c) { float d = row[c] - mean; var += d * d; }
      var *= (1.f / HIDDEN);
      float inv = rsqrtf(var + 1e-5f);
      for (int c = 0; c < HIDDEN; ++c)
        Hbuf[tid * HIDDEN + c] = (bf16)((row[c] - mean) * inv * lng[c] + lnb[c]);
    }
    __syncthreads();
    {
      v8f acc[2] = {};
      gemm_tile<4, 2>(Hbuf, HIDDEN, We, HIDDEN, wave * 32, acc);
      const int lane = tid & 31, n = lane & 15, mh = (lane >> 4) * 8;
#pragma unroll
      for (int ct = 0; ct < 2; ++ct) {
        int col = wave * 32 + ct * 16 + n;
#pragma unroll
        for (int r = 0; r < 8; ++r)
          hE[(size_t)(e0 + mh + r) * HIDDEN + col] = acc[ct][r] + bWe[col];
      }
    }
    __syncthreads();
  }
}

// ------------- fused per-node edge-MLP (message / dec-message / edge-upd) --
// MODE 0: encoder message (mask_attend, masked sum into msg_out)
// MODE 1: decoder message (plain sum into msg_out)
// MODE 2: encoder edge update (h_E = LN(n3, h_E + out))
template <int KA, int MODE>
__global__ __launch_bounds__(128) void edge_mlp_kernel(
    const float* __restrict__ hV, float* __restrict__ hE,
    const int* __restrict__ Eidx, const float* __restrict__ mm,
    const bf16* __restrict__ W1, const float* __restrict__ b1,
    const bf16* __restrict__ W2, const float* __restrict__ b2,
    const bf16* __restrict__ W3, const float* __restrict__ b3,
    const float* __restrict__ n3g, const float* __restrict__ n3b,
    float* __restrict__ msg_out) {
  const int node = blockIdx.x;
  const int jb = (node >> 11) << 11;
  const int tid = threadIdx.x, wave = tid >> 5;
  __shared__ bf16 Abuf[16 * KA];
  __shared__ bf16 Hbuf[16 * HIDDEN];
  __shared__ bf16 H2buf[16 * HIDDEN];
  __shared__ float Obuf[16 * HIDDEN];
  __shared__ float hVi[HIDDEN];
  __shared__ float attn[16];
  __shared__ float acc_sum[HIDDEN];
  if (tid < HIDDEN) {
    hVi[tid] = hV[(size_t)node * HIDDEN + tid];
    acc_sum[tid] = 0.f;
  }
  __syncthreads();
  const float mi = mm[node];
  for (int mt = 0; mt < 3; ++mt) {
    const int e0 = node * KNB + mt * 16;
    if (tid < 16) {
      if (MODE == 0) {
        int jg = jb + Eidx[e0 + tid];
        attn[tid] = mi * mm[jg];
      } else {
        attn[tid] = 1.f;
      }
    }
    for (int idx = tid; idx < 16 * KA; idx += 128) {
      int r = idx / KA, cc = idx % KA;
      float v;
      if (KA == 512) {
        if (cc < 256) v = hVi[cc & 127];
        else if (cc < 384) v = hE[(size_t)(e0 + r) * HIDDEN + (cc - 256)];
        else { int jg = jb + Eidx[e0 + r]; v = hV[(size_t)jg * HIDDEN + (cc - 384)]; }
      } else {
        if (cc < 128) v = hVi[cc];
        else if (cc < 256) v = hE[(size_t)(e0 + r) * HIDDEN + (cc - 128)];
        else { int jg = jb + Eidx[e0 + r]; v = hV[(size_t)jg * HIDDEN + (cc - 256)]; }
      }
      Abuf[idx] = (bf16)v;
    }
    __syncthreads();
    {
      v8f acc[2] = {};
      gemm_tile<KA / 32, 2>(Abuf, KA, W1, HIDDEN, wave * 32, acc);
      epi_bf16_gelu(Hbuf, HIDDEN, wave * 32 + 0, b1, acc[0]);
      epi_bf16_gelu(Hbuf, HIDDEN, wave * 32 + 16, b1, acc[1]);
    }
    __syncthreads();
    {
      v8f acc[2] = {};
      gemm_tile<4, 2>(Hbuf, HIDDEN, W2, HIDDEN, wave * 32, acc);
      epi_bf16_gelu(H2buf, HIDDEN, wave * 32 + 0, b2, acc[0]);
      epi_bf16_gelu(H2buf, HIDDEN, wave * 32 + 16, b2, acc[1]);
    }
    __syncthreads();
    {
      v8f acc[2] = {};
      gemm_tile<4, 2>(H2buf, HIDDEN, W3, HIDDEN, wave * 32, acc);
      if (MODE == 2) {
        epi_f32(Obuf, HIDDEN, wave * 32 + 0, b3, acc[0]);
        epi_f32(Obuf, HIDDEN, wave * 32 + 16, b3, acc[1]);
      } else {
        const int lane = tid & 31, n = lane & 15, mh = (lane >> 4) * 8;
#pragma unroll
        for (int ct = 0; ct < 2; ++ct) {
          int col = wave * 32 + ct * 16 + n;
          float s = 0.f;
#pragma unroll
          for (int r = 0; r < 8; ++r)
            s += (acc[ct][r] + b3[col]) * attn[mh + r];
          atomicAdd(&acc_sum[col], s);
        }
      }
    }
    __syncthreads();
    if (MODE == 2) {
      if (tid < 16) {
        const int e = e0 + tid;
        float mean = 0.f;
        for (int c = 0; c < HIDDEN; ++c)
          mean += hE[(size_t)e * HIDDEN + c] + Obuf[tid * HIDDEN + c];
        mean *= (1.f / HIDDEN);
        float var = 0.f;
        for (int c = 0; c < HIDDEN; ++c) {
          float v = hE[(size_t)e * HIDDEN + c] + Obuf[tid * HIDDEN + c] - mean;
          var += v * v;
        }
        var *= (1.f / HIDDEN);
        float inv = rsqrtf(var + 1e-5f);
        for (int c = 0; c < HIDDEN; ++c) {
          float v = hE[(size_t)e * HIDDEN + c] + Obuf[tid * HIDDEN + c];
          hE[(size_t)e * HIDDEN + c] = (v - mean) * inv * n3g[c] + n3b[c];
        }
      }
      __syncthreads();
    }
  }
  if (MODE != 2) {
    if (tid < HIDDEN) msg_out[(size_t)node * HIDDEN + tid] = acc_sum[tid];
  }
}

// --------------------- node residual + layernorm (n1) ----------------------
__global__ __launch_bounds__(128) void node_ln_kernel(
    float* __restrict__ hV, const float* __restrict__ msg,
    const float* __restrict__ g, const float* __restrict__ b, float scale_inv) {
  const int node = blockIdx.x;
  const int t = threadIdx.x;
  __shared__ float red[HIDDEN];
  float x = hV[(size_t)node * HIDDEN + t] + msg[(size_t)node * HIDDEN + t] * scale_inv;
  red[t] = x;
  __syncthreads();
  for (int s = 64; s > 0; s >>= 1) {
    if (t < s) red[t] += red[t + s];
    __syncthreads();
  }
  float mean = red[0] * (1.f / HIDDEN);
  __syncthreads();
  red[t] = (x - mean) * (x - mean);
  __syncthreads();
  for (int s = 64; s > 0; s >>= 1) {
    if (t < s) red[t] += red[t + s];
    __syncthreads();
  }
  float var = red[0] * (1.f / HIDDEN);
  hV[(size_t)node * HIDDEN + t] = (x - mean) * rsqrtf(var + 1e-5f) * g[t] + b[t];
}

// -------------------- node FFN: LN(n2, x + Wout(gelu(Win x))) * mask -------
__global__ __launch_bounds__(128) void ffn_kernel(
    float* __restrict__ hV,
    const bf16* __restrict__ Win, const float* __restrict__ binb,
    const bf16* __restrict__ Wout, const float* __restrict__ boutb,
    const float* __restrict__ n2g, const float* __restrict__ n2b,
    const float* __restrict__ mask) {
  const int base = blockIdx.x * 16;  // 16 nodes per block
  const int tid = threadIdx.x, wave = tid >> 5;
  __shared__ bf16 Xbuf[16 * HIDDEN];
  __shared__ bf16 Hbuf[16 * 512];
  __shared__ float Obuf[16 * HIDDEN];
  for (int idx = tid; idx < 16 * HIDDEN; idx += 128) {
    int r = idx >> 7, c = idx & 127;
    Xbuf[idx] = (bf16)hV[(size_t)(base + r) * HIDDEN + c];
  }
  __syncthreads();
  {
    v8f acc[8] = {};
    gemm_tile<4, 8>(Xbuf, HIDDEN, Win, 512, wave * 128, acc);
#pragma unroll
    for (int ct = 0; ct < 8; ++ct)
      epi_bf16_gelu(Hbuf, 512, wave * 128 + ct * 16, binb, acc[ct]);
  }
  __syncthreads();
  {
    v8f acc[2] = {};
    gemm_tile<16, 2>(Hbuf, 512, Wout, HIDDEN, wave * 32, acc);
    epi_f32(Obuf, HIDDEN, wave * 32 + 0, boutb, acc[0]);
    epi_f32(Obuf, HIDDEN, wave * 32 + 16, boutb, acc[1]);
  }
  __syncthreads();
  if (tid < 16) {
    const int node = base + tid;
    float mean = 0.f;
    for (int c = 0; c < HIDDEN; ++c)
      mean += hV[(size_t)node * HIDDEN + c] + Obuf[tid * HIDDEN + c];
    mean *= (1.f / HIDDEN);
    float var = 0.f;
    for (int c = 0; c < HIDDEN; ++c) {
      float v = hV[(size_t)node * HIDDEN + c] + Obuf[tid * HIDDEN + c] - mean;
      var += v * v;
    }
    var *= (1.f / HIDDEN);
    float inv = rsqrtf(var + 1e-5f);
    float mk = mask[node];
    for (int c = 0; c < HIDDEN; ++c) {
      float v = hV[(size_t)node * HIDDEN + c] + Obuf[tid * HIDDEN + c];
      hV[(size_t)node * HIDDEN + c] =
          ((v - mean) * inv * n2g[c] + n2b[c]) * mk;
    }
  }
}

// ---------------------- final head: latent -> (z, mean, logvar) ------------
__global__ __launch_bounds__(32) void final_kernel(
    const float* __restrict__ hV,
    const float* __restrict__ fw, const float* __restrict__ fb,
    const float* __restrict__ mw, const float* __restrict__ mb,
    const float* __restrict__ lw, const float* __restrict__ lb,
    const float* __restrict__ mm, const float* __restrict__ rmask,
    float* __restrict__ out) {
  const int node = blockIdx.x;
  const int t = threadIdx.x;  // 32
  __shared__ float lat[LATENT];
  float s = fb[t];
  for (int c = 0; c < HIDDEN; ++c)
    s += hV[(size_t)node * HIDDEN + c] * fw[c * LATENT + t];
  lat[t] = fmaxf(s, 0.f);
  __syncthreads();
  float me = mb[t], lv = lb[t];
  for (int c = 0; c < LATENT; ++c) {
    me += lat[c] * mw[c * LATENT + t];
    lv += lat[c] * lw[c * LATENT + t];
  }
  const float m = mm[node], rk = rmask[node];
  me *= m;
  lv *= m;
  unsigned id = (unsigned)(node * LATENT + t);
  float u1 = ((float)pcg_hash(id * 2u + 1u) + 1.0f) * (1.0f / 4294967296.0f);
  float u2 = (float)pcg_hash(id * 2u + 2u) * (1.0f / 4294967296.0f);
  float eps = sqrtf(-2.f * __logf(u1)) * __cosf(6.28318530717958647f * u2);
  float z = (me + eps * __expf(0.5f * lv)) * m;
  z *= rk; me *= rk; lv *= rk;
  const size_t NL = (size_t)NNODE * LATENT;
  out[(size_t)node * LATENT + t] = z;
  out[NL + (size_t)node * LATENT + t] = me;
  out[2 * NL + (size_t)node * LATENT + t] = lv;
}

// =============================== host side =================================
struct EncP {
  const float *W1b,*W1w,*W11b,*W11w,*W12b,*W12w,*W13b,*W13w,*W2b,*W2w,*W3b,*W3w,
              *Winb,*Winw,*Woutb,*Woutw,*n1b,*n1g,*n2b,*n2g,*n3b,*n3g;
};
struct DecP {
  const float *W1b,*W1w,*W2b,*W2w,*W3b,*W3w,*Winb,*Winw,*Woutb,*Woutw,
              *n1b,*n1g,*n2b,*n2g;
};
struct EncB { bf16 *W1,*W11,*W12,*W13,*W2,*W3,*Win,*Wout; };
struct DecB { bf16 *W1,*W2,*W3,*Win,*Wout; };

extern "C" void kernel_launch(void* const* d_in, const int* in_sizes, int n_in,
                              void* d_out, int out_size, void* d_ws,
                              size_t ws_size, hipStream_t stream) {
  (void)in_sizes; (void)n_in; (void)out_size; (void)ws_size;
  // ------- parse inputs (params pytree flattened, keys sorted) -------
  int ci = 0;
  const float* trans = (const float*)d_in[ci++];
  const float* rot   = (const float*)d_in[ci++];
  auto F = [&]() { return (const float*)d_in[ci++]; };
  const float* We_b = F(); const float* We_w = F();
  DecP dec[3];
  for (int l = 0; l < 3; ++l) {
    DecP& p = dec[l];
    p.W1b=F(); p.W1w=F(); p.W2b=F(); p.W2w=F(); p.W3b=F(); p.W3w=F();
    p.Winb=F(); p.Winw=F(); p.Woutb=F(); p.Woutw=F();
    p.n1b=F(); p.n1g=F(); p.n2b=F(); p.n2g=F();
  }
  const float* emb_b = F(); const float* emb_w = F();
  const float* en_b = F();  const float* en_g = F();
  EncP enc[3];
  for (int l = 0; l < 3; ++l) {
    EncP& p = enc[l];
    p.W1b=F(); p.W1w=F(); p.W11b=F(); p.W11w=F(); p.W12b=F(); p.W12w=F();
    p.W13b=F(); p.W13w=F(); p.W2b=F(); p.W2w=F(); p.W3b=F(); p.W3w=F();
    p.Winb=F(); p.Winw=F(); p.Woutb=F(); p.Woutw=F();
    p.n1b=F(); p.n1g=F(); p.n2b=F(); p.n2g=F(); p.n3b=F(); p.n3g=F();
  }
  const float* fcl_b = F(); const float* fcl_w = F();
  const float* fcm_b = F(); const float* fcm_w = F();
  const float* fin_b = F(); const float* fin_w = F();
  const float* pos_b = F(); const float* pos_w = F();
  const float* mm    = (const float*)d_in[ci++];
  const float* rmask = (const float*)d_in[ci++];
  ci++;  // aatype unused
  const int* resi  = (const int*)d_in[ci++];
  const int* chain = (const int*)d_in[ci++];

  // ------- workspace layout (bump allocator, ~107 MB) -------
  char* w = (char*)d_ws;
  size_t off = 0;
  auto alloc = [&](size_t bytes) {
    void* p = w + off;
    off += (bytes + 255) & ~(size_t)255;
    return p;
  };
  float* atoms = (float*)alloc((size_t)NNODE * 15 * 4);
  int* ridx    = (int*)alloc((size_t)NNODE * 4);
  int* cidx    = (int*)alloc((size_t)NNODE * 4);
  int* Eidx    = (int*)alloc((size_t)NNODE * KNB * 4);
  float* hE    = (float*)alloc((size_t)NNODE * KNB * HIDDEN * 4);
  float* hV    = (float*)alloc((size_t)NNODE * HIDDEN * 4);
  float* msg   = (float*)alloc((size_t)NNODE * HIDDEN * 4);
  auto ab = [&](int n) { return (bf16*)alloc((size_t)n * 2); };
  bf16* We16  = ab(HIDDEN * HIDDEN);
  bf16* emb16 = ab(416 * HIDDEN);
  EncB encB[3];
  for (int l = 0; l < 3; ++l) {
    encB[l].W1  = ab(384 * HIDDEN); encB[l].W11 = ab(384 * HIDDEN);
    encB[l].W12 = ab(HIDDEN * HIDDEN); encB[l].W13 = ab(HIDDEN * HIDDEN);
    encB[l].W2  = ab(HIDDEN * HIDDEN); encB[l].W3  = ab(HIDDEN * HIDDEN);
    encB[l].Win = ab(HIDDEN * 512); encB[l].Wout = ab(512 * HIDDEN);
  }
  DecB decB[3];
  for (int l = 0; l < 3; ++l) {
    decB[l].W1 = ab(512 * HIDDEN);
    decB[l].W2 = ab(HIDDEN * HIDDEN); decB[l].W3 = ab(HIDDEN * HIDDEN);
    decB[l].Win = ab(HIDDEN * 512); decB[l].Wout = ab(512 * HIDDEN);
  }

  auto cvt = [&](const float* s, bf16* d, int n) {
    cvt_f32_bf16_kernel<<<(n + 255) / 256, 256, 0, stream>>>(s, d, n);
  };
  cvt(We_w, We16, HIDDEN * HIDDEN);
  cvt(emb_w, emb16, 416 * HIDDEN);
  for (int l = 0; l < 3; ++l) {
    cvt(enc[l].W1w, encB[l].W1, 384 * HIDDEN);
    cvt(enc[l].W11w, encB[l].W11, 384 * HIDDEN);
    cvt(enc[l].W12w, encB[l].W12, HIDDEN * HIDDEN);
    cvt(enc[l].W13w, encB[l].W13, HIDDEN * HIDDEN);
    cvt(enc[l].W2w, encB[l].W2, HIDDEN * HIDDEN);
    cvt(enc[l].W3w, encB[l].W3, HIDDEN * HIDDEN);
    cvt(enc[l].Winw, encB[l].Win, HIDDEN * 512);
    cvt(enc[l].Woutw, encB[l].Wout, 512 * HIDDEN);
    cvt(dec[l].W1w, decB[l].W1, 512 * HIDDEN);
    cvt(dec[l].W2w, decB[l].W2, HIDDEN * HIDDEN);
    cvt(dec[l].W3w, decB[l].W3, HIDDEN * HIDDEN);
    cvt(dec[l].Winw, decB[l].Win, HIDDEN * 512);
    cvt(dec[l].Woutw, decB[l].Wout, 512 * HIDDEN);
  }

  // ------- pipeline -------
  atoms_kernel<<<(NNODE + 127) / 128, 128, 0, stream>>>(
      trans, rot, mm, rmask, resi, chain, atoms, ridx, cidx);
  topk_kernel<<<NNODE, 256, 0, stream>>>(atoms, mm, Eidx);
  edge_init_kernel<<<NNODE, 128, 0, stream>>>(
      atoms, ridx, cidx, Eidx, pos_w, pos_b, emb16, emb_b, en_g, en_b,
      We16, We_b, hE);
  zero_f32_kernel<<<(NNODE * HIDDEN + 255) / 256, 256, 0, stream>>>(
      hV, NNODE * HIDDEN);

  const float inv_scale = 1.0f / 30.0f;
  for (int l = 0; l < 3; ++l) {
    edge_mlp_kernel<384, 0><<<NNODE, 128, 0, stream>>>(
        hV, hE, Eidx, mm, encB[l].W1, enc[l].W1b, encB[l].W2, enc[l].W2b,
        encB[l].W3, enc[l].W3b, nullptr, nullptr, msg);
    node_ln_kernel<<<NNODE, 128, 0, stream>>>(hV, msg, enc[l].n1g, enc[l].n1b,
                                              inv_scale);
    ffn_kernel<<<NNODE / 16, 128, 0, stream>>>(
        hV, encB[l].Win, enc[l].Winb, encB[l].Wout, enc[l].Woutb,
        enc[l].n2g, enc[l].n2b, mm);
    edge_mlp_kernel<384, 2><<<NNODE, 128, 0, stream>>>(
        hV, hE, Eidx, mm, encB[l].W11, enc[l].W11b, encB[l].W12, enc[l].W12b,
        encB[l].W13, enc[l].W13b, enc[l].n3g, enc[l].n3b, msg);
  }
  for (int l = 0; l < 3; ++l) {
    edge_mlp_kernel<512, 1><<<NNODE, 128, 0, stream>>>(
        hV, hE, Eidx, mm, decB[l].W1, dec[l].W1b, decB[l].W2, dec[l].W2b,
        decB[l].W3, dec[l].W3b, nullptr, nullptr, msg);
    node_ln_kernel<<<NNODE, 128, 0, stream>>>(hV, msg, dec[l].n1g, dec[l].n1b,
                                              inv_scale);
    ffn_kernel<<<NNODE / 16, 128, 0, stream>>>(
        hV, decB[l].Win, dec[l].Winb, decB[l].Wout, dec[l].Woutb,
        dec[l].n2g, dec[l].n2b, rmask);
  }
  final_kernel<<<NNODE, 32, 0, stream>>>(hV, fin_w, fin_b, fcm_w, fcm_b,
                                         fcl_w, fcl_b, mm, rmask,
                                         (float*)d_out);
}
